// BERT_NER_45140106281695
// MI455X (gfx1250) — matrix operand order repaired and checked by reading, not compiled, not run on hardware
//
#include <hip/hip_runtime.h>
#include <hip/hip_bf16.h>

// ---------------------------------------------------------------------------
// BERT-NER CRF head for MI455X (gfx1250, wave32).
//  K1: logits = relu(hidden @ W + b)   via v_wmma_f32_16x16x32_f16
//  K2: CRF forward scan (log-likelihood per sequence), 1 wave / sequence
//  K3: Viterbi forward + backtrace (backpointers in LDS), 1 wave / sequence
//  K4: loss = -mean(ll)
// ---------------------------------------------------------------------------

#define Bsz   64
#define Tlen  512
#define Hdim  768
#define Ldim  9
#define LP    16          // L padded to 16 for WMMA N-dim and row stride
#define Mrows (Bsz * Tlen)
#define NEG_INF (-1.0e30f)

typedef __attribute__((ext_vector_type(16))) _Float16 v16h;
typedef __attribute__((ext_vector_type(8)))  float    v8f;
typedef __attribute__((ext_vector_type(2)))  _Float16 h2;

// ------------------------- K1: GEMM + bias + ReLU --------------------------
// Grid: 256 blocks x 256 threads (8 waves). Each wave computes one 16x16
// output tile (16 rows of logits, 16 cols of which 9 are real) over K=768
// in 24 steps of v_wmma_f32_16x16x32_f16.
__global__ __launch_bounds__(256) void gemm_relu_kernel(
    const float* __restrict__ hidden,   // [Mrows][Hdim]
    const float* __restrict__ W,        // [Hdim][Ldim]
    const float* __restrict__ bias,     // [Ldim]
    float* __restrict__ logits)         // [Mrows][LP] (ws)
{
    // Stage W transposed into LDS as f16: ldsWT[n][k], zero-padded n=9..15.
    __shared__ _Float16 ldsWT[LP * Hdim];   // 24 KB
    const int tid = threadIdx.x;
    for (int idx = tid; idx < LP * Hdim; idx += 256) {
        int n = idx & (LP - 1);
        int k = idx >> 4;
        float v = (n < Ldim) ? W[k * Ldim + n] : 0.0f;
        ldsWT[n * Hdim + k] = (_Float16)v;
    }
    __syncthreads();

    const int wave  = tid >> 5;
    const int lane  = tid & 31;
    const int mtile = blockIdx.x * 8 + wave;
    const int khalf = lane >> 4;        // 0: K-low half, 1: K-high half
    const int ncol  = lane & 15;        // B/C/D column, A row

    const float*    Arow = hidden + (size_t)((mtile << 4) + ncol) * Hdim;
    const _Float16* Wn   = ldsWT + ncol * Hdim;

    v8f acc = {};
    #pragma unroll 4
    for (int ks = 0; ks < Hdim / 32; ++ks) {
        const int kbase = ks * 32;
        // prefetch next K-slab of this row (speculative; OOB dropped by HW)
        __builtin_prefetch(Arow + kbase + 32, 0, 0);

        v16h a, bf;
        #pragma unroll
        for (int v = 0; v < 8; ++v) {
            // A 16x32 f16 layout: VGPR v<4 -> K = khalf*8 + 2v (+1)
            //                     VGPR v>=4 -> K = 16 + khalf*8 + 2(v-4) (+1)
            const int klocal = (v < 4) ? (khalf * 8 + 2 * v)
                                       : (16 + khalf * 8 + 2 * (v - 4));
            float2 f = *(const float2*)(Arow + kbase + klocal);
            a[2 * v]     = (_Float16)f.x;
            a[2 * v + 1] = (_Float16)f.y;

            // B 32x16 f16 layout: same K as A (klocal), N = ncol
            const int kb = kbase + klocal;
            h2 p = *(const h2*)(Wn + kb);
            bf[2 * v]     = p.x;
            bf[2 * v + 1] = p.y;
        }
        acc = __builtin_amdgcn_wmma_f32_16x16x32_f16(
                  false, a, false, bf, (short)0, acc, false, false);
    }

    // Epilogue: D layout f32 16x16 -> lane holds column ncol, rows v + 8*khalf
    if (ncol < Ldim) {
        const float bb = bias[ncol];
        #pragma unroll
        for (int v = 0; v < 8; ++v) {
            const int m = (mtile << 4) + v + 8 * khalf;
            float val = acc[v] + bb;
            logits[(size_t)m * LP + ncol] = val > 0.0f ? val : 0.0f;
        }
    }
}

// ------------------------- K2: CRF forward scan -----------------------------
// One wave32 per sequence. Lane j (j<9) holds alpha[j]; transitions column
// trans[:, j] lives in registers; 9-way logsumexp via __shfl.
__global__ __launch_bounds__(32) void crf_forward_kernel(
    const float* __restrict__ logits,       // [Mrows][LP]
    const float* __restrict__ trans,        // [Ldim][Ldim]
    const int*   __restrict__ tags,         // [Bsz][Tlen]
    const int*   __restrict__ lengths,      // [Bsz]
    float*       __restrict__ ll)           // [Bsz]
{
    const int b    = blockIdx.x;
    const int lane = threadIdx.x;
    const int len  = lengths[b];
    const float* lg = logits + (size_t)b * Tlen * LP;
    const int*   tg = tags + b * Tlen;

    float tr[Ldim];
    #pragma unroll
    for (int i = 0; i < Ldim; ++i)
        tr[i] = (lane < Ldim) ? trans[i * Ldim + lane] : 0.0f;

    float alpha = (lane < Ldim) ? lg[lane] : NEG_INF;
    float unary = 0.0f, binary = 0.0f;
    int prev_tag = tg[0];
    if (lane == 0) unary = lg[prev_tag];

    for (int t = 1; t < Tlen; ++t) {
        if (t < len) {
            float vals[Ldim];
            float m = NEG_INF;
            #pragma unroll
            for (int i = 0; i < Ldim; ++i) {
                float ai = __shfl(alpha, i, 32);
                float v  = ai + tr[i];
                vals[i]  = v;
                m = fmaxf(m, v);
            }
            float s = 0.0f;
            #pragma unroll
            for (int i = 0; i < Ldim; ++i) s += expf(vals[i] - m);
            float na = m + logf(s) + ((lane < Ldim) ? lg[t * LP + lane] : 0.0f);
            if (lane < Ldim) alpha = na;

            if (lane == 0) {
                const int ct = tg[t];
                unary  += lg[t * LP + ct];
                binary += trans[prev_tag * Ldim + ct];
                prev_tag = ct;
            }
        }
    }

    // log_norm = logsumexp over lanes 0..8
    float m = (lane < Ldim) ? alpha : NEG_INF;
    #pragma unroll
    for (int off = 16; off; off >>= 1) m = fmaxf(m, __shfl_xor(m, off, 32));
    float e = (lane < Ldim) ? expf(alpha - m) : 0.0f;
    #pragma unroll
    for (int off = 16; off; off >>= 1) e += __shfl_xor(e, off, 32);

    if (lane == 0) ll[b] = unary + binary - (m + logf(e));
}

// ------------------------- K3: Viterbi decode -------------------------------
// One wave32 per sequence. Backpointers kept in LDS (512x16 bytes = 8KB) so
// the serial backtrace stays on-chip. Decode written as float, masked to 0.
__global__ __launch_bounds__(32) void viterbi_kernel(
    const float* __restrict__ logits,       // [Mrows][LP]
    const float* __restrict__ trans,        // [Ldim][Ldim]
    const int*   __restrict__ lengths,      // [Bsz]
    float*       __restrict__ decode_out)   // [Bsz][Tlen] (floats)
{
    __shared__ unsigned char bp[Tlen * LP];

    const int b    = blockIdx.x;
    const int lane = threadIdx.x;
    const int len  = lengths[b];
    const float* lg = logits + (size_t)b * Tlen * LP;
    float* dec = decode_out + (size_t)b * Tlen;

    float tr[Ldim];
    #pragma unroll
    for (int i = 0; i < Ldim; ++i)
        tr[i] = (lane < Ldim) ? trans[i * Ldim + lane] : 0.0f;

    float state = (lane < Ldim) ? lg[lane] : NEG_INF;

    for (int t = 1; t < Tlen; ++t) {
        int bpi = lane;                      // identity when masked
        if (t < len) {
            float best = NEG_INF;
            int   bi   = 0;
            #pragma unroll
            for (int i = 0; i < Ldim; ++i) { // ascending i + strict > => first max
                float si = __shfl(state, i, 32);
                float v  = si + tr[i];
                if (v > best) { best = v; bi = i; }
            }
            if (lane < Ldim) {
                state = best + lg[t * LP + lane];
                bpi = bi;
            }
        }
        if (lane < LP) bp[t * LP + lane] = (unsigned char)bpi;
    }
    __syncthreads();

    // last_tag = first-occurrence argmax over state lanes 0..8
    float m = (lane < Ldim) ? state : NEG_INF;
    #pragma unroll
    for (int off = 16; off; off >>= 1) m = fmaxf(m, __shfl_xor(m, off, 32));
    unsigned long long hit = __ballot(lane < Ldim && state == m);
    int last = (int)(__ffsll((long long)hit) - 1);

    if (lane == 0) {
        int cur = last;
        for (int t = Tlen - 1; t >= 1; --t) {
            dec[t] = (t < len) ? (float)cur : 0.0f;
            cur = (int)bp[t * LP + cur];
        }
        dec[0] = (float)cur;                 // t=0 always valid (len >= 1)
    }
}

// ------------------------- K4: loss = -mean(ll) -----------------------------
__global__ __launch_bounds__(32) void loss_kernel(
    const float* __restrict__ ll, float* __restrict__ out)
{
    if (threadIdx.x == 0) {
        float s = 0.0f;
        for (int i = 0; i < Bsz; ++i) s += ll[i];
        out[0] = -s / (float)Bsz;
    }
}

// ---------------------------------------------------------------------------
extern "C" void kernel_launch(void* const* d_in, const int* in_sizes, int n_in,
                              void* d_out, int out_size, void* d_ws, size_t ws_size,
                              hipStream_t stream) {
    const float* hidden  = (const float*)d_in[0];   // [64,512,768]
    const float* W       = (const float*)d_in[1];   // [768,9]
    const float* bias    = (const float*)d_in[2];   // [9]
    const float* trans   = (const float*)d_in[3];   // [9,9]
    const int*   tags    = (const int*)d_in[4];     // [64,512]
    const int*   lengths = (const int*)d_in[5];     // [64]

    float* out = (float*)d_out;                     // [0]=loss, [1..]=decode

    float* logits = (float*)d_ws;                   // Mrows*LP floats = 2 MB
    float* ll     = logits + (size_t)Mrows * LP;    // 64 floats

    // K1: 2048 M-tiles / 8 waves per block = 256 blocks
    gemm_relu_kernel<<<dim3(Mrows / (16 * 8)), dim3(256), 0, stream>>>(
        hidden, W, bias, logits);

    // K2: one wave per sequence
    crf_forward_kernel<<<dim3(Bsz), dim3(32), 0, stream>>>(
        logits, trans, tags, lengths, ll);

    // K3: one wave per sequence (independent of K2; both read logits)
    viterbi_kernel<<<dim3(Bsz), dim3(32), 0, stream>>>(
        logits, trans, lengths, out + 1);

    // K4: final loss reduction
    loss_kernel<<<dim3(1), dim3(32), 0, stream>>>(ll, out);
}
